// AtnPool_26585847562620
// MI455X (gfx1250) — matrix operand order repaired and checked by reading, not compiled, not run on hardware
//
#include <hip/hip_runtime.h>
#include <math.h>

#define BB 32
#define SS 2048
#define DD 1024
#define HH 8
#define EE 32     // d_head
#define OO 128    // d_head_out
#define SC 128    // S-chunk rows

typedef __bf16        v16bf __attribute__((ext_vector_type(16)));
typedef float         v8f   __attribute__((ext_vector_type(8)));
typedef unsigned int  uv4   __attribute__((ext_vector_type(4)));
typedef unsigned int  uv2   __attribute__((ext_vector_type(2)));
typedef float         fv4   __attribute__((ext_vector_type(4)));

union Frag { uv4 q[2]; v16bf v; };
union FU   { float f; unsigned u; };

__device__ __forceinline__ unsigned f2bf_bits(float f) {
  FU x; x.f = f;
  unsigned r = x.u + 0x7FFFu + ((x.u >> 16) & 1u);   // round-to-nearest-even
  return r >> 16;
}

// pack two floats to two truncated bf16s in one v_perm_b32
__device__ __forceinline__ unsigned pack_bf16_trunc(float lo, float hi) {
  FU a; a.f = lo;
  FU c; c.f = hi;
  return __builtin_amdgcn_perm(c.u, a.u, 0x07060302u);
}

__global__ __launch_bounds__(256)
void atnpool_kernel(const float* __restrict__ feat, const float* __restrict__ w1,
                    const float* __restrict__ b1,   const float* __restrict__ w2,
                    const float* __restrict__ b2,   float* __restrict__ out)
{
  // Fragment-ordered weight caches + staging buffers (~131 KB => 2 WGs / 320KB WGP)
  __shared__ unsigned short w1f[32 * 2 * 32 * 16];   // [kstep(32)][ntile(2)][lane(32)][16]  64KB
  __shared__ unsigned short w2f[8 * 32 * 16];        // [ntile(8)][lane(32)][16]              8KB
  __shared__ unsigned short Aslab[SC * 128];         // features slab bf16                   32KB
  __shared__ unsigned short h1s[SC * EE];            // GELU(h1) chunk bf16                   8KB
  __shared__ float pA[16 * OO];                      // per-(wave,half) partials              8KB
  __shared__ float pB[16 * OO];                      //                                       8KB
  __shared__ float m_run[OO], z_run[OO], acc_run[OO], newm[OO];
  __shared__ float b1s[EE], b2s[OO];

  const int tid    = threadIdx.x;
  const int wave   = tid >> 5;
  const int lane   = tid & 31;
  const int lanelo = lane & 15;
  const int half   = lane >> 4;
  const int b      = blockIdx.x / HH;
  const int h      = blockIdx.x % HH;

  // ---- pack w1[h] into WMMA B-fragment order (bf16, RNE, one-time) ----
  for (int s = tid; s < 32 * 2 * 32; s += 256) {
    int ks = s >> 6;
    int n  = (s >> 5) & 1;
    int ln = s & 31;
    int hb = ln >> 4;
    int e  = n * 16 + (ln & 15);
    const float* wp = w1 + ((size_t)h * DD + ks * 32) * EE + e;
    unsigned short* dst = &w1f[(size_t)s * 16];
    for (int j = 0; j < 16; ++j)
      dst[j] = (unsigned short)f2bf_bits(wp[(size_t)(8 * hb + (j & 7) + 16 * (j >> 3)) * EE]);
  }
  // ---- pack w2[h] ----
  for (int s = tid; s < 8 * 32; s += 256) {
    int n  = s >> 5;
    int ln = s & 31;
    int hb = ln >> 4;
    int e  = n * 16 + (ln & 15);
    const float* wp = w2 + ((size_t)h * EE) * OO + e;
    unsigned short* dst = &w2f[(size_t)s * 16];
    for (int j = 0; j < 16; ++j)
      dst[j] = (unsigned short)f2bf_bits(wp[(size_t)(8 * hb + (j & 7) + 16 * (j >> 3)) * OO]);
  }
  if (tid < EE) b1s[tid] = b1[h * EE + tid];
  if (tid < OO) b2s[tid] = b2[h * OO + tid];
  if (tid < OO) { m_run[tid] = -3.0e38f; z_run[tid] = 0.f; acc_run[tid] = 0.f; }
  __syncthreads();

  for (int chunk = 0; chunk < SS / SC; ++chunk) {
    const int s0 = chunk * SC;
    v8f c0, c1;
    for (int i = 0; i < 8; ++i) { c0[i] = 0.f; c1[i] = 0.f; }

    // ================= GEMM1: h = feat(128x1024) x w1 (1024x32) =================
    for (int ks8 = 0; ks8 < 8; ++ks8) {
      // stage 128x128 feature slab (f32 -> bf16, one v_perm per 2 elems)
      for (int i = 0; i < 16; ++i) {
        int idx = tid + 256 * i;
        int row = idx >> 5;
        int c4  = idx & 31;
        fv4 v = *reinterpret_cast<const fv4*>(
            &feat[((size_t)(b * SS + s0 + row)) * DD + ks8 * 128 + c4 * 4]);
        uv2 pk;
        pk.x = pack_bf16_trunc(v.x, v.y);
        pk.y = pack_bf16_trunc(v.z, v.w);
        *reinterpret_cast<uv2*>(&Aslab[row * 128 + c4 * 4]) = pk;
      }
      // prefetch next slab into L2 while this one is consumed (global_prefetch_b8)
      {
        int nks = ks8 + 1;
        int ns0 = s0;
        if (nks == 8) { nks = 0; ns0 = s0 + SC; }
        if (ns0 < SS) {
          int row = tid >> 1;
          int col = (tid & 1) * 64;
          __builtin_prefetch(&feat[((size_t)(b * SS + ns0 + row)) * DD + nks * 128 + col], 0, 1);
        }
      }
      __syncthreads();
      for (int kk = 0; kk < 4; ++kk) {
        Frag a;
        int m = wave * 16 + lanelo;
        a.q[0] = *reinterpret_cast<const uv4*>(&Aslab[m * 128 + kk * 32 + 8 * half]);
        a.q[1] = *reinterpret_cast<const uv4*>(&Aslab[m * 128 + kk * 32 + 16 + 8 * half]);
        int ksg = ks8 * 4 + kk;
        Frag bw0, bw1;
        const uv4* p0 = reinterpret_cast<const uv4*>(&w1f[(((ksg * 2 + 0) * 32) + lane) * 16]);
        bw0.q[0] = p0[0]; bw0.q[1] = p0[1];
        const uv4* p1 = reinterpret_cast<const uv4*>(&w1f[(((ksg * 2 + 1) * 32) + lane) * 16]);
        bw1.q[0] = p1[0]; bw1.q[1] = p1[1];
        c0 = __builtin_amdgcn_wmma_f32_16x16x32_bf16(false, a.v, false, bw0.v, (short)0, c0, false, false);
        c1 = __builtin_amdgcn_wmma_f32_16x16x32_bf16(false, a.v, false, bw1.v, (short)0, c1, false, false);
      }
      __syncthreads();
    }

    // ================= bias + exact GELU, write tile for GEMM2 =================
    for (int r = 0; r < 8; ++r) {
      int rowLocal = r + 8 * half;
      int srow = wave * 16 + rowLocal;
      float x0 = c0[r] + b1s[lanelo];
      float x1 = c1[r] + b1s[16 + lanelo];
      float g0 = 0.5f * x0 * (1.f + erff(x0 * 0.70710678118f));
      float g1 = 0.5f * x1 * (1.f + erff(x1 * 0.70710678118f));
      h1s[srow * EE + lanelo]      = (unsigned short)f2bf_bits(g0);
      h1s[srow * EE + 16 + lanelo] = (unsigned short)f2bf_bits(g1);
    }
    __syncthreads();

    // ================= GEMM2: L = gelu(h)(128x32) x w2 (32x128) =================
    Frag a2;
    {
      int m = wave * 16 + lanelo;
      a2.q[0] = *reinterpret_cast<const uv4*>(&h1s[m * EE + 8 * half]);
      a2.q[1] = *reinterpret_cast<const uv4*>(&h1s[m * EE + 16 + 8 * half]);
    }
    v8f L[8];
    for (int n = 0; n < 8; ++n) {
      Frag bw;
      const uv4* p = reinterpret_cast<const uv4*>(&w2f[((n * 32) + lane) * 16]);
      bw.q[0] = p[0]; bw.q[1] = p[1];
      v8f cz; for (int i = 0; i < 8; ++i) cz[i] = 0.f;
      L[n] = __builtin_amdgcn_wmma_f32_16x16x32_bf16(false, a2.v, false, bw.v, (short)0, cz, false, false);
      float bb = b2s[n * 16 + lanelo];
      float mx = -3.0e38f;
      for (int r = 0; r < 8; ++r) { L[n][r] += bb; mx = fmaxf(mx, L[n][r]); }
      pA[(wave * 2 + half) * OO + n * 16 + lanelo] = mx;   // per-column partial max
    }
    __syncthreads();

    // ================= flash-softmax: rescale running state =================
    if (tid < OO) {
      float cm = -3.0e38f;
      for (int i = 0; i < 16; ++i) cm = fmaxf(cm, pA[i * OO + tid]);
      float nm = fmaxf(m_run[tid], cm);
      float sc = __expf(m_run[tid] - nm);
      z_run[tid]   *= sc;
      acc_run[tid] *= sc;
      m_run[tid] = nm;
      newm[tid]  = nm;
    }
    __syncthreads();

    // ================= exp + feature-weighted accumulation =================
    for (int n = 0; n < 8; ++n) {
      int o = n * 16 + lanelo;
      float nm = newm[o];
      float zp = 0.f, ap = 0.f;
      for (int r = 0; r < 8; ++r) {
        float p = __expf(L[n][r] - nm);
        zp += p;
        int srow = s0 + wave * 16 + r + 8 * half;
        ap += p * feat[((size_t)(b * SS + srow)) * DD + h * OO + o];
      }
      pA[(wave * 2 + half) * OO + o] = zp;
      pB[(wave * 2 + half) * OO + o] = ap;
    }
    __syncthreads();
    if (tid < OO) {
      float zs = 0.f, as = 0.f;
      for (int i = 0; i < 16; ++i) { zs += pA[i * OO + tid]; as += pB[i * OO + tid]; }
      z_run[tid]   += zs;
      acc_run[tid] += as;
    }
    __syncthreads();
  }

  if (tid < OO)
    out[(size_t)b * DD + h * OO + tid] = acc_run[tid] / z_run[tid];
}

extern "C" void kernel_launch(void* const* d_in, const int* in_sizes, int n_in,
                              void* d_out, int out_size, void* d_ws, size_t ws_size,
                              hipStream_t stream) {
  (void)in_sizes; (void)n_in; (void)d_ws; (void)ws_size; (void)out_size;
  const float* feat = (const float*)d_in[0];
  const float* w1   = (const float*)d_in[1];
  const float* b1   = (const float*)d_in[2];
  const float* w2   = (const float*)d_in[3];
  const float* b2   = (const float*)d_in[4];
  float* out        = (float*)d_out;
  dim3 grid(BB * HH);
  dim3 block(256);
  hipLaunchKernelGGL(atnpool_kernel, grid, block, 0, stream, feat, w1, b1, w2, b2, out);
}
